// GCN_7069516169348
// MI455X (gfx1250) — matrix operand order repaired and checked by reading, not compiled, not run on hardware
//
#include <hip/hip_runtime.h>
#include <hip/hip_bf16.h>
#include <math.h>

typedef __bf16 bf16_t;
typedef __attribute__((ext_vector_type(16))) __bf16 v16bf;
typedef __attribute__((ext_vector_type(8)))  float  v8f;

#define F_IN  128
#define F_HID 256
#define F_OUT 16

// ---------------- degree / norm ----------------

__global__ void k_deg_init(float* __restrict__ deg, int n) {
  int i = blockIdx.x * blockDim.x + threadIdx.x;
  if (i < n) deg[i] = 1.0f;  // self-loop
}

__global__ void k_deg_accum(const long long* __restrict__ ei, float* __restrict__ deg, int E) {
  int e = blockIdx.x * blockDim.x + threadIdx.x;
  if (e < E) atomicAdd(&deg[(int)ei[(size_t)E + e]], 1.0f);
}

__global__ void k_rsqrt(float* __restrict__ d, int n) {
  int i = blockIdx.x * blockDim.x + threadIdx.x;
  if (i < n) d[i] = rsqrtf(d[i]);
}

// ---------------- GEMM1: h1 = x @ W1 (bf16 WMMA, f32 acc) ----------------
// one wave per 16x16 tile; epilogue also seeds agg1 with the self-loop term.

__global__ void k_gemm1(const float* __restrict__ x, const float* __restrict__ W1,
                        const float* __restrict__ dinv,
                        float* __restrict__ h1, float* __restrict__ agg1,
                        int mtiles) {
  const int gwave = (blockIdx.x * blockDim.x + threadIdx.x) >> 5;
  const int lane  = threadIdx.x & 31;
  const int mt = gwave >> 4;      // row tile
  const int nt = gwave & 15;      // 256/16 = 16 col tiles
  if (mt >= mtiles) return;       // wave-uniform -> EXEC all ones for WMMA

  const int row  = lane & 15;
  const int half = lane >> 4;
  const float* xrow = x + (size_t)(mt * 16 + row) * F_IN;

  v8f acc = {};
#pragma unroll
  for (int k0 = 0; k0 < F_IN; k0 += 32) {
    v16bf a, b;
    // A fragment: 16x32 bf16. lane<16: K = k0+{0..7, 16..23}; lane>=16: +8.
    const float* pa = xrow + k0 + half * 8;
#pragma unroll
    for (int g = 0; g < 8; ++g) {
      a[g]     = (bf16_t)pa[g];
      a[8 + g] = (bf16_t)pa[16 + g];
    }
    // B fragment: 32x16. col = lane&15, K = k0+{0..7, 16..23}; lane>=16: +8.
    const int colW = nt * 16 + row;
    const int kb   = half * 8;
#pragma unroll
    for (int j = 0; j < 16; ++j)
      b[j] = (bf16_t)W1[(size_t)(k0 + kb + (j < 8 ? j : 8 + j)) * F_HID + colW];

    acc = __builtin_amdgcn_wmma_f32_16x16x32_bf16(false, a, false, b,
                                                  (short)0, acc, false, false);
  }

#pragma unroll
  for (int r = 0; r < 8; ++r) {
    const int m = mt * 16 + half * 8 + r;
    const int n = nt * 16 + row;
    const float v = acc[r];
    const size_t idx = (size_t)m * F_HID + n;
    h1[idx] = v;
    const float di = dinv[m];
    agg1[idx] = v * di * di;   // self-loop contribution of segment_sum
  }
}

// ---------------- edge scatter, layer 1 ----------------
// 64 consecutive lanes per edge, float4 per lane -> coalesced 1KB gather/scatter.

__global__ void k_scatter1(const long long* __restrict__ ei,
                           const float* __restrict__ dinv,
                           const float* __restrict__ h1,
                           float* __restrict__ agg1, int E) {
  int t = blockIdx.x * blockDim.x + threadIdx.x;
  if (t >= E * 64) return;
  const int e = t >> 6;
  const int c = t & 63;
  const int src = (int)ei[e];
  const int dst = (int)ei[(size_t)E + e];
  const float norm = dinv[src] * dinv[dst];
  const float4 v = *(const float4*)(h1 + (size_t)src * F_HID + c * 4);
  float* o = agg1 + (size_t)dst * F_HID + c * 4;
  atomicAdd(o + 0, v.x * norm);
  atomicAdd(o + 1, v.y * norm);
  atomicAdd(o + 2, v.z * norm);
  atomicAdd(o + 3, v.w * norm);
}

// ---------------- +b1, ReLU, cvt bf16 (overwrites the dead h1 region) ----------------

__global__ void k_finalize1(const float* __restrict__ agg1, const float* __restrict__ b1,
                            bf16_t* __restrict__ relu1, int total) {
  int t = blockIdx.x * blockDim.x + threadIdx.x;
  if (t >= total) return;
  const int f = t & (F_HID - 1);
  relu1[t] = (bf16_t)fmaxf(agg1[t] + b1[f], 0.0f);
}

// ---------------- GEMM2: h2 = relu1 @ W2, seeds d_out with self-loop term ----------------

__global__ void k_gemm2(const bf16_t* __restrict__ relu1, const float* __restrict__ W2,
                        const float* __restrict__ dinv,
                        float* __restrict__ h2, float* __restrict__ outInit,
                        int mtiles) {
  const int mt   = (blockIdx.x * blockDim.x + threadIdx.x) >> 5;  // single N tile
  const int lane = threadIdx.x & 31;
  if (mt >= mtiles) return;  // wave-uniform

  const int row  = lane & 15;
  const int half = lane >> 4;
  const bf16_t* arow = relu1 + (size_t)(mt * 16 + row) * F_HID;

  v8f acc = {};
#pragma unroll
  for (int k0 = 0; k0 < F_HID; k0 += 32) {
    v16bf a, b;
    const bf16_t* pa = arow + k0 + half * 8;
#pragma unroll
    for (int g = 0; g < 8; ++g) {
      a[g]     = pa[g];
      a[8 + g] = pa[16 + g];
    }
    const int kb = half * 8;
#pragma unroll
    for (int j = 0; j < 16; ++j)
      b[j] = (bf16_t)W2[(size_t)(k0 + kb + (j < 8 ? j : 8 + j)) * F_OUT + row];

    acc = __builtin_amdgcn_wmma_f32_16x16x32_bf16(false, a, false, b,
                                                  (short)0, acc, false, false);
  }

#pragma unroll
  for (int r = 0; r < 8; ++r) {
    const int m = mt * 16 + half * 8 + r;
    const int n = row;
    const float v = acc[r];
    const size_t idx = (size_t)m * F_OUT + n;
    h2[idx] = v;
    const float di = dinv[m];
    outInit[idx] = v * di * di;
  }
}

// ---------------- edge scatter, layer 2 ----------------

__global__ void k_scatter2(const long long* __restrict__ ei,
                           const float* __restrict__ dinv,
                           const float* __restrict__ h2,
                           float* __restrict__ out, int E) {
  int t = blockIdx.x * blockDim.x + threadIdx.x;
  if (t >= E * 4) return;
  const int e = t >> 2;
  const int c = t & 3;
  const int src = (int)ei[e];
  const int dst = (int)ei[(size_t)E + e];
  const float norm = dinv[src] * dinv[dst];
  const float4 v = *(const float4*)(h2 + (size_t)src * F_OUT + c * 4);
  float* o = out + (size_t)dst * F_OUT + c * 4;
  atomicAdd(o + 0, v.x * norm);
  atomicAdd(o + 1, v.y * norm);
  atomicAdd(o + 2, v.z * norm);
  atomicAdd(o + 3, v.w * norm);
}

// ---------------- +b2, log_softmax (in place, one thread per row) ----------------

__global__ void k_finalize2(float* __restrict__ out, const float* __restrict__ b2, int n) {
  int i = blockIdx.x * blockDim.x + threadIdx.x;
  if (i >= n) return;
  float v[F_OUT];
  float mx = -INFINITY;
#pragma unroll
  for (int j = 0; j < F_OUT; ++j) {
    v[j] = out[(size_t)i * F_OUT + j] + b2[j];
    mx = fmaxf(mx, v[j]);
  }
  float s = 0.0f;
#pragma unroll
  for (int j = 0; j < F_OUT; ++j) s += expf(v[j] - mx);
  const float ls = logf(s) + mx;
#pragma unroll
  for (int j = 0; j < F_OUT; ++j) out[(size_t)i * F_OUT + j] = v[j] - ls;
}

// ---------------- launch ----------------

extern "C" void kernel_launch(void* const* d_in, const int* in_sizes, int n_in,
                              void* d_out, int out_size, void* d_ws, size_t ws_size,
                              hipStream_t stream) {
  (void)n_in; (void)out_size; (void)ws_size;
  const float*     x  = (const float*)d_in[0];
  const long long* ei = (const long long*)d_in[1];
  const float*     W1 = (const float*)d_in[2];
  const float*     b1 = (const float*)d_in[3];
  const float*     W2 = (const float*)d_in[4];
  const float*     b2 = (const float*)d_in[5];
  float* out = (float*)d_out;

  const int Nn = in_sizes[0] / F_IN;   // 100000
  const int E  = in_sizes[1] / 2;      // 1600000

  // workspace layout (all offsets 256B aligned):
  //   [0, 1MB)                  : deg -> dinv (N floats)
  //   [1MB, 1MB+102.4MB)        : h1 f32; later reused as relu1 bf16
  //   [+102.4MB, +204.8MB)      : agg1 f32
  //   [+204.8MB, +211.2MB)      : h2 f32
  char* ws = (char*)d_ws;
  const size_t szH1 = (size_t)Nn * F_HID * sizeof(float);
  float*  dinv  = (float*)ws;
  float*  h1    = (float*)(ws + (1u << 20));
  float*  agg1  = (float*)(ws + (1u << 20) + szH1);
  float*  h2    = (float*)(ws + (1u << 20) + 2 * szH1);
  bf16_t* relu1 = (bf16_t*)h1;   // h1 dead after scatter1; reuse region

  const int T = 256;
  k_deg_init <<<(Nn + T - 1) / T, T, 0, stream>>>(dinv, Nn);
  k_deg_accum<<<(E  + T - 1) / T, T, 0, stream>>>(ei, dinv, E);
  k_rsqrt    <<<(Nn + T - 1) / T, T, 0, stream>>>(dinv, Nn);

  const int mtiles = (Nn + 15) / 16;   // 6250
  {
    const int waves = mtiles * 16;     // 16 N-tiles
    k_gemm1<<<(waves * 32 + T - 1) / T, T, 0, stream>>>(x, W1, dinv, h1, agg1, mtiles);
  }
  k_scatter1<<<(E * 64 + T - 1) / T, T, 0, stream>>>(ei, dinv, h1, agg1, E);
  {
    const int total = Nn * F_HID;
    k_finalize1<<<(total + T - 1) / T, T, 0, stream>>>(agg1, b1, relu1, total);
  }
  k_gemm2<<<(mtiles * 32 + T - 1) / T, T, 0, stream>>>(relu1, W2, dinv, h2, out, mtiles);
  k_scatter2<<<(E * 4 + T - 1) / T, T, 0, stream>>>(ei, dinv, h2, out, E);
  k_finalize2<<<(Nn + T - 1) / T, T, 0, stream>>>(out, b2, Nn);
}
